// ISA_13460427505759
// MI455X (gfx1250) — matrix operand order repaired and checked by reading, not compiled, not run on hardware
//
#include <hip/hip_runtime.h>
#include <hip/hip_bf16.h>
#include <math.h>

#define BB 8
#define NN 1369
#define DIN 768
#define SS 7
#define DD 256
#define NITERS 3
#define RESG 37
#define SIGMAF 5.0f
#define EPSV 1e-8f
#define MROWS (BB*NN)            /* 10952 */
#define MTILES ((MROWS+15)/16)   /* 685   */
#define MPAD (MTILES*16)         /* 10960 */
#define NTILES ((NN+15)/16)      /* 86    */
#define SCALEF 0.0625f           /* 256^-0.5 */

typedef __bf16 bf16;
typedef __attribute__((ext_vector_type(16))) __bf16 bf16x16;
typedef __attribute__((ext_vector_type(8)))  __bf16 bf16x8;
typedef __attribute__((ext_vector_type(8)))  float  f32x8;

// ---------------------------------------------------------------- WMMA core
__device__ __forceinline__ f32x8 wmma_bf16(bf16x16 a, bf16x16 b, f32x8 c) {
  return __builtin_amdgcn_wmma_f32_16x16x32_bf16(
      /*neg_a=*/false, a, /*neg_b=*/false, b,
      /*c_mod=*/(short)0, c, /*reuse_a=*/false, /*reuse_b=*/false);
}

// A tile loader: row-major bf16 [16 x K], ISA 16-bit A layout.
// lane<16: row=lane, K = {0..7,16..23}+32*kc ; lane>=16: row=lane-16, K = {8..15,24..31}+32*kc
__device__ __forceinline__ bf16x16 load_a16(const bf16* base, int ldk, int kc, int lane) {
  int r  = lane & 15;
  int kh = (lane >> 4) << 3;                 // 0 or 8
  const bf16* p = base + (size_t)r * ldk + kc * 32 + kh;
  bf16x8 c0 = *(const bf16x8*)p;
  bf16x8 c1 = *(const bf16x8*)(p + 16);
  bf16x16 a;
#pragma unroll
  for (int e = 0; e < 8; ++e) { a[e] = c0[e]; a[e + 8] = c1[e]; }
  return a;
}

// B tile loader from pre-shuffled weights: Wp[(kc*NT+nt)*32 + lane] is a 32B packet
__device__ __forceinline__ bf16x16 load_bp(const bf16* Wp, int NT, int kc, int nt, int lane) {
  return ((const bf16x16*)Wp)[(size_t)(kc * NT + nt) * 32 + lane];
}

__device__ __forceinline__ float block_reduce_sum(float v, float* sbuf) {
  int tid = threadIdx.x;
  sbuf[tid] = v;
  __syncthreads();
#pragma unroll
  for (int off = 128; off > 0; off >>= 1) {
    if (tid < off) sbuf[tid] += sbuf[tid + off];
    __syncthreads();
  }
  float r = sbuf[0];
  __syncthreads();
  return r;
}

// ---------------------------------------------------------------- weight shuffle
// Pack fp32 W[K][Ncol] into bf16 WMMA B-layout tiles.
__global__ void pack_b_kernel(const float* __restrict__ W, bf16* __restrict__ Wp,
                              int K, int Ncol) {
  int id = blockIdx.x * blockDim.x + threadIdx.x;
  if (id >= K * Ncol) return;
  int e  = id & 15;
  int l  = (id >> 4) & 31;
  int t  = id >> 9;                 // tile = kc*NT + nt
  int NT = Ncol >> 4;
  int kc = t / NT, nt = t % NT;
  int kk = kc * 32 + ((l >> 4) << 3) + (e & 7) + ((e >> 3) << 4);   // K index
  int nn = nt * 16 + (l & 15);              // N index
  Wp[id] = (bf16)W[(size_t)kk * Ncol + nn];
}

// ---------------------------------------------------------------- generic GEMM
// out[M,Ncol] = act(A[M,K] @ B + bias). A bf16 row-major, B packed bf16.
// 8 waves/block, each wave = one 16x16 column tile; grid = (MTILES, Ncol/128).
__global__ void gemm_bf16_kernel(const bf16* __restrict__ A, const bf16* __restrict__ Bp,
                                 const float* __restrict__ bias,
                                 float* __restrict__ outF, bf16* __restrict__ outH,
                                 int K, int Ncol, int do_relu) {
  int lane = threadIdx.x & 31;
  int wave = threadIdx.x >> 5;
  int mt   = blockIdx.x;
  int nt   = blockIdx.y * 8 + wave;
  int NT   = Ncol >> 4;
  int col  = nt * 16 + (lane & 15);
  float bv = bias ? bias[col] : 0.0f;
  f32x8 acc;
#pragma unroll
  for (int i = 0; i < 8; ++i) acc[i] = bv;
  const bf16* arow = A + (size_t)mt * 16 * K;
  int nk = K >> 5;
  for (int kc = 0; kc < nk; ++kc) {
    bf16x16 a = load_a16(arow, K, kc, lane);
    bf16x16 b = load_bp(Bp, NT, kc, nt, lane);
    acc = wmma_bf16(a, b, acc);
  }
  int rb = mt * 16 + ((lane >> 4) << 3);
#pragma unroll
  for (int i = 0; i < 8; ++i) {
    float v = acc[i];
    if (do_relu) v = fmaxf(v, 0.0f);
    size_t idx = (size_t)(rb + i) * Ncol + col;
    if (outF) outF[idx] = v; else outH[idx] = (bf16)v;
  }
}

// ---------------------------------------------------------------- layernorms
__global__ void ln1_kernel(const float* __restrict__ x, const float* __restrict__ g,
                           const float* __restrict__ b, bf16* __restrict__ out, int rows) {
  __shared__ float sbuf[256];
  int row = blockIdx.x, tid = threadIdx.x;
  if (row >= rows) {
#pragma unroll
    for (int j = 0; j < 3; ++j) out[(size_t)row * DIN + tid + 256 * j] = (bf16)0.0f;
    return;
  }
  const float* xr = x + (size_t)row * DIN;
  float a0 = xr[tid], a1 = xr[tid + 256], a2 = xr[tid + 512];
  float mean = block_reduce_sum(a0 + a1 + a2, sbuf) * (1.0f / DIN);
  float d0 = a0 - mean, d1 = a1 - mean, d2 = a2 - mean;
  float var = block_reduce_sum(d0 * d0 + d1 * d1 + d2 * d2, sbuf) * (1.0f / DIN);
  float rs = rsqrtf(var + 1e-5f);
  out[(size_t)row * DIN + tid      ] = (bf16)(d0 * rs * g[tid      ] + b[tid      ]);
  out[(size_t)row * DIN + tid + 256] = (bf16)(d1 * rs * g[tid + 256] + b[tid + 256]);
  out[(size_t)row * DIN + tid + 512] = (bf16)(d2 * rs * g[tid + 512] + b[tid + 512]);
}

__global__ void ln2_kernel(const float* __restrict__ x, const float* __restrict__ g,
                           const float* __restrict__ b, bf16* __restrict__ out, int rows) {
  __shared__ float sbuf[256];
  int row = blockIdx.x, tid = threadIdx.x;
  if (row >= rows) { out[(size_t)row * DD + tid] = (bf16)0.0f; return; }
  float a = x[(size_t)row * DD + tid];
  float mean = block_reduce_sum(a, sbuf) * (1.0f / DD);
  float d = a - mean;
  float var = block_reduce_sum(d * d, sbuf) * (1.0f / DD);
  out[(size_t)row * DD + tid] = (bf16)(d * rsqrtf(var + 1e-5f) * g[tid] + b[tid]);
}

// ---------------------------------------------------------------- init
__global__ void init_kernel(const float* __restrict__ slots_init_p,
                            const float* __restrict__ S_s0, const float* __restrict__ S_p0,
                            float* __restrict__ slots, float* __restrict__ Sp,
                            float* __restrict__ Ss, float* __restrict__ gxy) {
  int id = blockIdx.x * 256 + threadIdx.x;
  if (id < NN) {
    int ix = id % RESG, iy = id / RESG;
    gxy[id * 2    ] = -1.0f + 2.0f * (float)ix / (float)(RESG - 1);
    gxy[id * 2 + 1] = -1.0f + 2.0f * (float)iy / (float)(RESG - 1);
  }
  if (id < BB * SS * DD) {
    int s = (id / DD) % SS;
    slots[id] = slots_init_p[s * DD + (id % DD)];
  }
  if (id < BB * SS * 2) {
    int s = (id / 2) % SS;
    Sp[id] = S_p0[s * 2 + (id & 1)];
    Ss[id] = S_s0[s * 2 + (id & 1)];
  }
}

// ---------------------------------------------------------------- fused f_mlp tile
// For block (b,s,ntile): z = X + rel@g_w + g_b (bf16 in LDS), h = relu(z@fw1+b1),
// y = h@fw2+b2 (f32 in LDS). Ends with all yA visible.
__device__ void fmlp_tile(const float* __restrict__ XR, int b, int s, int ntile,
                          const float* __restrict__ Sp, const float* __restrict__ Ss,
                          const float* __restrict__ g_w, const float* __restrict__ g_b,
                          const bf16* __restrict__ fw1p, const float* __restrict__ fb1,
                          const bf16* __restrict__ fw2p, const float* __restrict__ fb2,
                          const float* __restrict__ gxy,
                          bf16* zA, bf16* hA, float* yA) {
  int tid = threadIdx.x, lane = tid & 31, wave = tid >> 5;
  int bs = b * SS + s;
  int r = tid >> 4, dsub = tid & 15;
  int n = ntile * 16 + r;
  float spx = Sp[bs * 2], spy = Sp[bs * 2 + 1];
  float ssx = Ss[bs * 2] * SIGMAF, ssy = Ss[bs * 2 + 1] * SIGMAF;
  if (n < NN) {
    float rel0 = (gxy[n * 2]     - spx) / ssx;
    float rel1 = (gxy[n * 2 + 1] - spy) / ssy;
    const float* xr = XR + ((size_t)b * NN + n) * DD;
#pragma unroll
    for (int j = 0; j < 16; ++j) {
      int d = dsub * 16 + j;
      zA[r * DD + d] = (bf16)(xr[d] + rel0 * g_w[d] + rel1 * g_w[DD + d] + g_b[d]);
    }
  } else {
#pragma unroll
    for (int j = 0; j < 16; ++j) zA[r * DD + dsub * 16 + j] = (bf16)0.0f;
  }
  __syncthreads();
  // GEMM1: h = relu(z @ fw1 + fb1), 16 col tiles over 8 waves
#pragma unroll
  for (int t2 = 0; t2 < 2; ++t2) {
    int nt = wave * 2 + t2;
    int coln = nt * 16 + (lane & 15);
    float bv = fb1[coln];
    f32x8 acc;
#pragma unroll
    for (int i = 0; i < 8; ++i) acc[i] = bv;
#pragma unroll
    for (int kc = 0; kc < 8; ++kc) {
      bf16x16 a = load_a16(zA, DD, kc, lane);
      bf16x16 bm = load_bp(fw1p, 16, kc, nt, lane);
      acc = wmma_bf16(a, bm, acc);
    }
    int rb = (lane >> 4) << 3;
#pragma unroll
    for (int i = 0; i < 8; ++i) hA[(rb + i) * DD + coln] = (bf16)fmaxf(acc[i], 0.0f);
  }
  __syncthreads();
  // GEMM2: y = h @ fw2 + fb2
#pragma unroll
  for (int t2 = 0; t2 < 2; ++t2) {
    int nt = wave * 2 + t2;
    int coln = nt * 16 + (lane & 15);
    float bv = fb2[coln];
    f32x8 acc;
#pragma unroll
    for (int i = 0; i < 8; ++i) acc[i] = bv;
#pragma unroll
    for (int kc = 0; kc < 8; ++kc) {
      bf16x16 a = load_a16(hA, DD, kc, lane);
      bf16x16 bm = load_bp(fw2p, 16, kc, nt, lane);
      acc = wmma_bf16(a, bm, acc);
    }
    int rb = (lane >> 4) << 3;
#pragma unroll
    for (int i = 0; i < 8; ++i) yA[(rb + i) * DD + coln] = acc[i];
  }
  __syncthreads();
}

// ---------------------------------------------------------------- pass 1: dots
__global__ void pass1_kernel(const float* __restrict__ KX, const float* __restrict__ q,
                             const float* __restrict__ Sp, const float* __restrict__ Ss,
                             const float* __restrict__ g_w, const float* __restrict__ g_b,
                             const bf16* __restrict__ fw1p, const float* __restrict__ fb1,
                             const bf16* __restrict__ fw2p, const float* __restrict__ fb2,
                             const float* __restrict__ gxy, float* __restrict__ dots) {
  __shared__ __align__(16) bf16 zA[16 * DD];
  __shared__ __align__(16) bf16 hA[16 * DD];
  __shared__ float yA[16 * DD];
  __shared__ float qs[DD];
  __shared__ float red[256];
  int ntile = blockIdx.x, s = blockIdx.y, b = blockIdx.z;
  int tid = threadIdx.x;
  int bs = b * SS + s;
  qs[tid] = q[bs * DD + tid];
  fmlp_tile(KX, b, s, ntile, Sp, Ss, g_w, g_b, fw1p, fb1, fw2p, fb2, gxy, zA, hA, yA);
  int r = tid >> 4, dsub = tid & 15;
  float part = 0.0f;
#pragma unroll
  for (int j = 0; j < 16; ++j) {
    int d = j * 16 + dsub;
    part += yA[r * DD + d] * qs[d];
  }
  red[tid] = part; __syncthreads();
  if (dsub < 8) red[tid] += red[tid + 8]; __syncthreads();
  if (dsub < 4) red[tid] += red[tid + 4]; __syncthreads();
  if (dsub < 2) red[tid] += red[tid + 2]; __syncthreads();
  int n = ntile * 16 + r;
  if (dsub == 0 && n < NN) dots[(size_t)bs * NN + n] = (red[tid] + red[tid + 1]) * SCALEF;
}

// ---------------------------------------------------------------- pass 2: updates
__global__ void pass2_kernel(const float* __restrict__ VX, const float* __restrict__ attn,
                             const float* __restrict__ Sp, const float* __restrict__ Ss,
                             const float* __restrict__ g_w, const float* __restrict__ g_b,
                             const bf16* __restrict__ fw1p, const float* __restrict__ fb1,
                             const bf16* __restrict__ fw2p, const float* __restrict__ fb2,
                             const float* __restrict__ gxy, float* __restrict__ updates) {
  __shared__ __align__(16) bf16 zA[16 * DD];
  __shared__ __align__(16) bf16 hA[16 * DD];
  __shared__ float yA[16 * DD];
  __shared__ float aw[16];
  int ntile = blockIdx.x, s = blockIdx.y, b = blockIdx.z;
  int tid = threadIdx.x;
  int bs = b * SS + s;
  if (tid < 16) {
    int n = ntile * 16 + tid;
    aw[tid] = (n < NN) ? attn[(size_t)bs * NN + n] : 0.0f;
  }
  fmlp_tile(VX, b, s, ntile, Sp, Ss, g_w, g_b, fw1p, fb1, fw2p, fb2, gxy, zA, hA, yA);
  float acc = 0.0f;
#pragma unroll
  for (int rr = 0; rr < 16; ++rr) acc += aw[rr] * yA[rr * DD + tid];
  atomicAdd(&updates[bs * DD + tid], acc);
}

// ---------------------------------------------------------------- q = LN(slots) @ Wq
__global__ void slotq_kernel(const float* __restrict__ slots, const float* __restrict__ g,
                             const float* __restrict__ b, const float* __restrict__ Wq,
                             float* __restrict__ q) {
  __shared__ float sn[DD];
  __shared__ float sbuf[256];
  int bs = blockIdx.x, d = threadIdx.x;
  float x = slots[bs * DD + d];
  float mean = block_reduce_sum(x, sbuf) * (1.0f / DD);
  float dx = x - mean;
  float var = block_reduce_sum(dx * dx, sbuf) * (1.0f / DD);
  sn[d] = dx * rsqrtf(var + 1e-5f) * g[d] + b[d];
  __syncthreads();
  float acc = 0.0f;
  for (int e = 0; e < DD; ++e) acc += sn[e] * Wq[e * DD + d];
  q[bs * DD + d] = acc;
}

// ---------------------------------------------------------------- softmax over slots
__global__ void softmax_kernel(const float* __restrict__ dots, float* __restrict__ attn_u) {
  int id = blockIdx.x * blockDim.x + threadIdx.x;
  if (id >= MROWS) return;
  int b = id / NN, n = id % NN;
  float v[SS];
  float mx = -3.0e38f;
#pragma unroll
  for (int s = 0; s < SS; ++s) { v[s] = dots[((size_t)b * SS + s) * NN + n]; mx = fmaxf(mx, v[s]); }
  float sum = 0.0f;
#pragma unroll
  for (int s = 0; s < SS; ++s) { v[s] = expf(v[s] - mx); sum += v[s]; }
  float inv = 1.0f / sum;
#pragma unroll
  for (int s = 0; s < SS; ++s) attn_u[((size_t)b * SS + s) * NN + n] = v[s] * inv + EPSV;
}

// -------------------------------------------- normalize over n; S_p, S_s; zero updates
__global__ void attn_norm_kernel(const float* __restrict__ attn_u, const float* __restrict__ gxy,
                                 float* __restrict__ attn, float* __restrict__ SpW,
                                 float* __restrict__ SsW, float* __restrict__ updates) {
  __shared__ float sbuf[256];
  int bs = blockIdx.x, tid = threadIdx.x;
  updates[bs * DD + tid] = 0.0f;
  float tot = 0.0f, sx = 0.0f, sy = 0.0f;
  for (int n = tid; n < NN; n += 256) {
    float a = attn_u[(size_t)bs * NN + n];
    tot += a; sx += a * gxy[n * 2]; sy += a * gxy[n * 2 + 1];
  }
  tot = block_reduce_sum(tot, sbuf);
  sx  = block_reduce_sum(sx, sbuf);
  sy  = block_reduce_sum(sy, sbuf);
  float inv = 1.0f / tot;
  float spx = sx * inv, spy = sy * inv;
  if (tid == 0) { SpW[bs * 2] = spx; SpW[bs * 2 + 1] = spy; }
  float vx = 0.0f, vy = 0.0f;
  for (int n = tid; n < NN; n += 256) {
    float a = attn_u[(size_t)bs * NN + n] * inv;
    attn[(size_t)bs * NN + n] = a;
    float dx = gxy[n * 2] - spx, dy = gxy[n * 2 + 1] - spy;
    vx += a * dx * dx; vy += a * dy * dy;
  }
  vx = block_reduce_sum(vx, sbuf);
  vy = block_reduce_sum(vy, sbuf);
  if (tid == 0) { SsW[bs * 2] = sqrtf(vx); SsW[bs * 2 + 1] = sqrtf(vy); }
}

// ---------------------------------------------------------------- GRU + residual MLP
__global__ void gru_mlp_kernel(const float* __restrict__ updates, const float* __restrict__ slots_in,
                               const float* __restrict__ wih, const float* __restrict__ whh,
                               const float* __restrict__ bih, const float* __restrict__ bhh,
                               const float* __restrict__ mg, const float* __restrict__ mb,
                               const float* __restrict__ w1, const float* __restrict__ b1,
                               const float* __restrict__ w2, const float* __restrict__ b2,
                               float* __restrict__ slots_out) {
  __shared__ float u[DD], h[DD], gi[3 * DD], gh[3 * DD], hn[DD], m[DD], hid[4 * DD];
  __shared__ float sbuf[256];
  int bs = blockIdx.x, d = threadIdx.x;
  u[d] = updates[bs * DD + d];
  h[d] = slots_in[bs * DD + d];
  __syncthreads();
  for (int j = d; j < 3 * DD; j += 256) {
    float a = bih[j], c = bhh[j];
    for (int e = 0; e < DD; ++e) { a += u[e] * wih[e * 3 * DD + j]; c += h[e] * whh[e * 3 * DD + j]; }
    gi[j] = a; gh[j] = c;
  }
  __syncthreads();
  float rg = 1.0f / (1.0f + expf(-(gi[d] + gh[d])));
  float zg = 1.0f / (1.0f + expf(-(gi[DD + d] + gh[DD + d])));
  float ng = tanhf(gi[2 * DD + d] + rg * gh[2 * DD + d]);
  float hnew = (1.0f - zg) * ng + zg * h[d];
  hn[d] = hnew;
  __syncthreads();
  float mean = block_reduce_sum(hnew, sbuf) * (1.0f / DD);
  float dx = hnew - mean;
  float var = block_reduce_sum(dx * dx, sbuf) * (1.0f / DD);
  m[d] = dx * rsqrtf(var + 1e-5f) * mg[d] + mb[d];
  __syncthreads();
  for (int j = d; j < 4 * DD; j += 256) {
    float a = b1[j];
    for (int e = 0; e < DD; ++e) a += m[e] * w1[e * 4 * DD + j];
    hid[j] = fmaxf(a, 0.0f);
  }
  __syncthreads();
  float o = b2[d];
  for (int j = 0; j < 4 * DD; ++j) o += hid[j] * w2[j * DD + d];
  slots_out[bs * DD + d] = hn[d] + o;
}

// ---------------------------------------------------------------- final projection
__global__ void final_out_kernel(const float* __restrict__ slots, const float* __restrict__ w,
                                 const float* __restrict__ b, float* __restrict__ out) {
  __shared__ float sl[DD];
  int bs = blockIdx.x, d = threadIdx.x;
  sl[d] = slots[bs * DD + d];
  __syncthreads();
  float a = b[d];
  for (int e = 0; e < DD; ++e) a += sl[e] * w[e * DD + d];
  out[bs * DD + d] = a;
}

// ================================================================ launcher
extern "C" void kernel_launch(void* const* d_in, const int* in_sizes, int n_in,
                              void* d_out, int out_size, void* d_ws, size_t ws_size,
                              hipStream_t stream) {
  const float* inputs       = (const float*)d_in[0];
  const float* slots_init_p = (const float*)d_in[1];
  const float* S_s0         = (const float*)d_in[2];
  const float* S_p0         = (const float*)d_in[3];
  const float* im_ln1_g     = (const float*)d_in[4];
  const float* im_ln1_b     = (const float*)d_in[5];
  const float* im_w1        = (const float*)d_in[6];
  const float* im_b1        = (const float*)d_in[7];
  const float* im_w2        = (const float*)d_in[8];
  const float* im_b2        = (const float*)d_in[9];
  const float* im_ln2_g     = (const float*)d_in[10];
  const float* im_ln2_b     = (const float*)d_in[11];
  const float* Wq           = (const float*)d_in[12];
  const float* Wk           = (const float*)d_in[13];
  const float* Wv           = (const float*)d_in[14];
  const float* g_w          = (const float*)d_in[15];
  const float* g_b          = (const float*)d_in[16];
  const float* f_w1         = (const float*)d_in[17];
  const float* f_b1         = (const float*)d_in[18];
  const float* f_w2         = (const float*)d_in[19];
  const float* f_b2         = (const float*)d_in[20];
  const float* ln_g         = (const float*)d_in[21];
  const float* ln_b         = (const float*)d_in[22];
  const float* gru_wih      = (const float*)d_in[23];
  const float* gru_whh      = (const float*)d_in[24];
  const float* gru_bih      = (const float*)d_in[25];
  const float* gru_bhh      = (const float*)d_in[26];
  const float* mlp_ln_g     = (const float*)d_in[27];
  const float* mlp_ln_b     = (const float*)d_in[28];
  const float* mlp_w1       = (const float*)d_in[29];
  const float* mlp_b1       = (const float*)d_in[30];
  const float* mlp_w2       = (const float*)d_in[31];
  const float* mlp_b2       = (const float*)d_in[32];
  const float* fin_w        = (const float*)d_in[33];
  const float* fin_b        = (const float*)d_in[34];

  size_t off = 0;
  auto carve = [&](size_t bytes) -> void* {
    void* p = (void*)((char*)d_ws + off);
    off += (bytes + 255) & ~(size_t)255;
    return p;
  };
  bf16*  Xln   = (bf16*) carve((size_t)MPAD * DIN * sizeof(bf16));
  bf16*  H1    = (bf16*) carve((size_t)MPAD * DIN * sizeof(bf16));
  float* X2    = (float*)carve((size_t)MPAD * DD * sizeof(float));
  bf16*  Xnb   = (bf16*) carve((size_t)MPAD * DD * sizeof(bf16));
  float* KX    = (float*)carve((size_t)MPAD * DD * sizeof(float));
  float* VX    = (float*)carve((size_t)MPAD * DD * sizeof(float));
  bf16*  w1p   = (bf16*) carve((size_t)DIN * DIN * sizeof(bf16));
  bf16*  w2p   = (bf16*) carve((size_t)DIN * DD * sizeof(bf16));
  bf16*  wkp   = (bf16*) carve((size_t)DD * DD * sizeof(bf16));
  bf16*  wvp   = (bf16*) carve((size_t)DD * DD * sizeof(bf16));
  bf16*  fw1p  = (bf16*) carve((size_t)DD * DD * sizeof(bf16));
  bf16*  fw2p  = (bf16*) carve((size_t)DD * DD * sizeof(bf16));
  float* dots  = (float*)carve((size_t)BB * SS * NN * sizeof(float));
  float* attnu = (float*)carve((size_t)BB * SS * NN * sizeof(float));
  float* slots = (float*)carve((size_t)BB * SS * DD * sizeof(float));
  float* qbuf  = (float*)carve((size_t)BB * SS * DD * sizeof(float));
  float* updts = (float*)carve((size_t)BB * SS * DD * sizeof(float));
  float* Sp0   = (float*)carve((size_t)BB * SS * 2 * sizeof(float));
  float* Sp1   = (float*)carve((size_t)BB * SS * 2 * sizeof(float));
  float* Ss0   = (float*)carve((size_t)BB * SS * 2 * sizeof(float));
  float* Ss1   = (float*)carve((size_t)BB * SS * 2 * sizeof(float));
  float* gxy   = (float*)carve((size_t)NN * 2 * sizeof(float));
  (void)in_sizes; (void)n_in; (void)out_size; (void)ws_size;

  float* out0 = (float*)d_out;
  float* attn = out0 + BB * SS * DD;

  // --- weight shuffles
  pack_b_kernel<<<(DIN * DIN + 255) / 256, 256, 0, stream>>>(im_w1, w1p, DIN, DIN);
  pack_b_kernel<<<(DIN * DD + 255) / 256, 256, 0, stream>>>(im_w2, w2p, DIN, DD);
  pack_b_kernel<<<(DD * DD + 255) / 256, 256, 0, stream>>>(Wk, wkp, DD, DD);
  pack_b_kernel<<<(DD * DD + 255) / 256, 256, 0, stream>>>(Wv, wvp, DD, DD);
  pack_b_kernel<<<(DD * DD + 255) / 256, 256, 0, stream>>>(f_w1, fw1p, DD, DD);
  pack_b_kernel<<<(DD * DD + 255) / 256, 256, 0, stream>>>(f_w2, fw2p, DD, DD);
  init_kernel<<<BB * SS, 256, 0, stream>>>(slots_init_p, S_s0, S_p0, slots, Sp0, Ss0, gxy);

  // --- encoder
  ln1_kernel<<<MPAD, 256, 0, stream>>>(inputs, im_ln1_g, im_ln1_b, Xln, MROWS);
  gemm_bf16_kernel<<<dim3(MTILES, DIN / 128), 256, 0, stream>>>(Xln, w1p, im_b1, nullptr, H1, DIN, DIN, 1);
  gemm_bf16_kernel<<<dim3(MTILES, DD / 128), 256, 0, stream>>>(H1, w2p, im_b2, X2, nullptr, DIN, DD, 0);
  ln2_kernel<<<MPAD, 256, 0, stream>>>(X2, im_ln2_g, im_ln2_b, Xnb, MROWS);
  gemm_bf16_kernel<<<dim3(MTILES, DD / 128), 256, 0, stream>>>(Xnb, wkp, nullptr, KX, nullptr, DD, DD, 0);
  gemm_bf16_kernel<<<dim3(MTILES, DD / 128), 256, 0, stream>>>(Xnb, wvp, nullptr, VX, nullptr, DD, DD, 0);

  // --- slot attention iterations (t = 0..ITERS; straight-through is fwd-identity)
  float* SpB[2] = {Sp0, Sp1};
  float* SsB[2] = {Ss0, Ss1};
  for (int t = 0; t <= NITERS; ++t) {
    const float* SpR = SpB[t & 1];
    const float* SsR = SsB[t & 1];
    float* SpW = SpB[1 - (t & 1)];
    float* SsW = SsB[1 - (t & 1)];
    slotq_kernel<<<BB * SS, 256, 0, stream>>>(slots, ln_g, ln_b, Wq, qbuf);
    pass1_kernel<<<dim3(NTILES, SS, BB), 256, 0, stream>>>(KX, qbuf, SpR, SsR, g_w, g_b,
                                                           fw1p, f_b1, fw2p, f_b2, gxy, dots);
    softmax_kernel<<<(MROWS + 255) / 256, 256, 0, stream>>>(dots, attnu);
    attn_norm_kernel<<<BB * SS, 256, 0, stream>>>(attnu, gxy, attn, SpW, SsW, updts);
    if (t < NITERS) {
      pass2_kernel<<<dim3(NTILES, SS, BB), 256, 0, stream>>>(VX, attn, SpR, SsR, g_w, g_b,
                                                             fw1p, f_b1, fw2p, f_b2, gxy, updts);
      gru_mlp_kernel<<<BB * SS, 256, 0, stream>>>(updts, slots, gru_wih, gru_whh, gru_bih, gru_bhh,
                                                  mlp_ln_g, mlp_ln_b, mlp_w1, mlp_b1, mlp_w2, mlp_b2,
                                                  slots);
    }
  }
  final_out_kernel<<<BB * SS, 256, 0, stream>>>(slots, fin_w, fin_b, out0);
}
